// GNP_Block_54597624267062
// MI455X (gfx1250) — matrix operand order repaired and checked by reading, not compiled, not run on hardware
//
#include <hip/hip_runtime.h>

#define Nn   100000
#define Ee   500000
#define ELG  1000000
#define Hh   128
#define KGEc 128
#define NG   1024
#define NITER 2
#define EPSL 1e-5f

typedef __attribute__((ext_vector_type(16))) _Float16 v16h;
typedef __attribute__((ext_vector_type(8)))  float    v8f;

__device__ __forceinline__ float sigmf(float x) { return 1.0f / (1.0f + __expf(-x)); }

// A fragment (16x32 f16, MxK): lane L holds row L%16; g = L/16.
// halves 0..7  -> K = kb*32 + 8g + i
// halves 8..15 -> K = kb*32 + 16 + 8g + (i-8)
__device__ __forceinline__ v16h load_a_f32(const float* __restrict__ Arow, int gsel, float scale) {
  v16h a;
  const float* p0 = Arow + 8 * gsel;
  const float* p1 = Arow + 16 + 8 * gsel;
#pragma unroll
  for (int i = 0; i < 8; ++i) a[i] = (_Float16)(p0[i] * scale);
#pragma unroll
  for (int i = 0; i < 8; ++i) a[8 + i] = (_Float16)(p1[i] * scale);
  return a;
}

// B fragment (32x16 f16, KxN), B[k][n] = W[n][k]: lane L holds col L%16;
// halves 0..7 -> K = kb*32 + 8g + i; halves 8..15 -> K = kb*32 + 16 + 8g + (i-8), along W row `col`.
__device__ __forceinline__ v16h load_b_h(const _Float16* __restrict__ Wrow, int gsel) {
  v16h b;
  const _Float16* p = Wrow + 8 * gsel;
#pragma unroll
  for (int i = 0; i < 8; ++i) { b[i] = p[i]; b[8 + i] = p[16 + i]; }
  return b;
}

__device__ __forceinline__ v8f wmma_f16(v16h a, v16h b, v8f c) {
  return __builtin_amdgcn_wmma_f32_16x16x32_f16(false, a, false, b, (short)0, c, false, false);
}

// ---------------- elementwise / scatter kernels ----------------

__global__ void k_cvt(const float* __restrict__ s, _Float16* __restrict__ d, int n) {
  int i = blockIdx.x * blockDim.x + threadIdx.x;
  if (i < n) d[i] = (_Float16)s[i];
}

__global__ void k_count(const int* __restrict__ idx, float* __restrict__ cnt, int n) {
  int i = blockIdx.x * blockDim.x + threadIdx.x;
  if (i < n) atomicAdd(&cnt[idx[i]], 1.0f);
}

// fused[e,:] = edge_attr[e,:] + 0.5*(x[src[e],:] + x[dst[e],:]); one wave per edge, float4 lanes
__global__ void k_fuse(const float* __restrict__ ea, const float* __restrict__ x,
                       const int* __restrict__ src, const int* __restrict__ dst,
                       float* __restrict__ fused) {
  int t = blockIdx.x * blockDim.x + threadIdx.x;
  int e = t >> 5;
  if (e >= Ee) return;
  int q = (t & 31) * 4;
  int s = src[e], d = dst[e];
  float4 xs = *(const float4*)(x + (size_t)s * Hh + q);
  float4 xd = *(const float4*)(x + (size_t)d * Hh + q);
  float4 ev = *(const float4*)(ea + (size_t)e * Hh + q);
  float4 r;
  r.x = ev.x + 0.5f * (xs.x + xd.x);
  r.y = ev.y + 0.5f * (xs.y + xd.y);
  r.z = ev.z + 0.5f * (xs.z + xd.z);
  r.w = ev.w + 0.5f * (xs.w + xd.w);
  *(float4*)(fused + (size_t)e * Hh + q) = r;
}

// dstmat[didx[i],:] += srcmat[(sidx?sidx[i]:i),:]   (float atomics, float4 source reads)
__global__ void k_scatter_add(const float* __restrict__ srcmat, const int* __restrict__ sidx,
                              const int* __restrict__ didx, float* __restrict__ dstmat, int nitems) {
  int t = blockIdx.x * blockDim.x + threadIdx.x;
  int i = t >> 5;
  if (i >= nitems) return;
  int q = (t & 31) * 4;
  int a = sidx ? sidx[i] : i;
  int b = didx[i];
  float4 v = *(const float4*)(srcmat + (size_t)a * Hh + q);
  float* p = dstmat + (size_t)b * Hh + q;
  atomicAdd(p + 0, v.x);
  atomicAdd(p + 1, v.y);
  atomicAdd(p + 2, v.z);
  atomicAdd(p + 3, v.w);
}

// ---------------- WMMA GEMM kernels ----------------

// fused += BN(PReLU((agg/deg) @ W1^T + b1));  one wave per 16-row tile, 8 col tiles, K=128
__global__ void k_edge_gemm(const float* __restrict__ agg, const float* __restrict__ deg,
                            const _Float16* __restrict__ W1h, const float* __restrict__ b1,
                            const float* __restrict__ a1, const float* __restrict__ gamma,
                            const float* __restrict__ beta, const float* __restrict__ mean,
                            const float* __restrict__ var, float* __restrict__ fused) {
  int wave = (blockIdx.x * blockDim.x + threadIdx.x) >> 5;
  int lane = threadIdx.x & 31;
  int row0 = wave << 4;
  if (row0 >= Ee) return;
  int lrow = row0 + (lane & 15);
  int gsel = lane >> 4;
  float scale = 1.0f / fmaxf(deg[lrow], 1.0f);  // fold seg-mean divide into A
  const float* Ar = agg + (size_t)lrow * Hh;
  v16h afr[4];
#pragma unroll
  for (int kb = 0; kb < 4; ++kb) afr[kb] = load_a_f32(Ar + kb * 32, gsel, scale);
  float alpha = a1[0];
  for (int ct = 0; ct < 8; ++ct) {
    int col = (ct << 4) + (lane & 15);
    v8f acc = {};
    const _Float16* Wrow = W1h + (size_t)col * Hh;
#pragma unroll
    for (int kb = 0; kb < 4; ++kb) acc = wmma_f16(afr[kb], load_b_h(Wrow + kb * 32, gsel), acc);
    float inv = rsqrtf(var[col] + EPSL);
    float ga = gamma[col] * inv;
    float bb = beta[col] - mean[col] * ga;
    float bias = b1[col];
    int rb = row0 + (gsel << 3);
#pragma unroll
    for (int i = 0; i < 8; ++i) {
      float v = acc[i] + bias;
      v = (v >= 0.0f) ? v : alpha * v;          // PReLU
      v = v * ga + bb;                          // BatchNorm (eval)
      fused[(size_t)(rb + i) * Hh + col] += v;  // residual
    }
  }
}

// Full GRU step fused: gi = (nupd/deg)@w_ih^T+b_ih, gh = h@w_hh^T+b_hh, gates, h in-place.
__global__ void k_gru(const float* __restrict__ nupd, const float* __restrict__ degn,
                      const _Float16* __restrict__ wih, const _Float16* __restrict__ whh,
                      const float* __restrict__ bih, const float* __restrict__ bhh,
                      float* __restrict__ h) {
  int wave = (blockIdx.x * blockDim.x + threadIdx.x) >> 5;
  int lane = threadIdx.x & 31;
  int row0 = wave << 4;
  if (row0 >= Nn) return;
  int lrow = row0 + (lane & 15);
  int gsel = lane >> 4;
  float scale = 1.0f / fmaxf(degn[lrow], 1.0f);
  const float* Ai = nupd + (size_t)lrow * Hh;
  const float* Ah = h + (size_t)lrow * Hh;
  v16h ai[4], ah[4];
#pragma unroll
  for (int kb = 0; kb < 4; ++kb) {
    ai[kb] = load_a_f32(Ai + kb * 32, gsel, scale);
    ah[kb] = load_a_f32(Ah + kb * 32, gsel, 1.0f);
  }
  for (int ct = 0; ct < 8; ++ct) {
    int col = (ct << 4) + (lane & 15);
    v8f ar = {}, az = {}, agi = {}, agh = {};
    const _Float16* wir = wih + (size_t)col * Hh;
    const _Float16* wiz = wih + (size_t)(col + Hh) * Hh;
    const _Float16* win = wih + (size_t)(col + 2 * Hh) * Hh;
    const _Float16* whr = whh + (size_t)col * Hh;
    const _Float16* whz = whh + (size_t)(col + Hh) * Hh;
    const _Float16* whn = whh + (size_t)(col + 2 * Hh) * Hh;
#pragma unroll
    for (int kb = 0; kb < 4; ++kb) {
      ar  = wmma_f16(ai[kb], load_b_h(wir + kb * 32, gsel), ar);   // gi_r
      ar  = wmma_f16(ah[kb], load_b_h(whr + kb * 32, gsel), ar);   // +gh_r
      az  = wmma_f16(ai[kb], load_b_h(wiz + kb * 32, gsel), az);   // gi_z
      az  = wmma_f16(ah[kb], load_b_h(whz + kb * 32, gsel), az);   // +gh_z
      agi = wmma_f16(ai[kb], load_b_h(win + kb * 32, gsel), agi);  // gi_n
      agh = wmma_f16(ah[kb], load_b_h(whn + kb * 32, gsel), agh);  // gh_n
    }
    float br_ = bih[col] + bhh[col];
    float bz_ = bih[col + Hh] + bhh[col + Hh];
    float bin_ = bih[col + 2 * Hh];
    float bhn_ = bhh[col + 2 * Hh];
    int rb = row0 + (gsel << 3);
#pragma unroll
    for (int i = 0; i < 8; ++i) {
      float rg = sigmf(ar[i] + br_);
      float zg = sigmf(az[i] + bz_);
      float ng = tanhf(agi[i] + bin_ + rg * (agh[i] + bhn_));
      size_t off = (size_t)(rb + i) * Hh + col;
      float hv = h[off];
      h[off] = (1.0f - zg) * ng + zg * hv;
    }
  }
}

// graph_emb = (attn[g]*gsum[g]) @ Wr^T + br
__global__ void k_emb(const float* __restrict__ gsum, const float* __restrict__ attn,
                      const _Float16* __restrict__ Wrh, const float* __restrict__ br,
                      float* __restrict__ out) {
  int wave = (blockIdx.x * blockDim.x + threadIdx.x) >> 5;
  int lane = threadIdx.x & 31;
  int row0 = wave << 4;
  if (row0 >= NG) return;
  int lrow = row0 + (lane & 15);
  int gsel = lane >> 4;
  float scale = attn[lrow];  // attended sum == attn * segment_sum(x)
  const float* Ar = gsum + (size_t)lrow * Hh;
  v16h afr[4];
#pragma unroll
  for (int kb = 0; kb < 4; ++kb) afr[kb] = load_a_f32(Ar + kb * 32, gsel, scale);
  for (int ct = 0; ct < 8; ++ct) {
    int col = (ct << 4) + (lane & 15);
    v8f acc = {};
    const _Float16* Wrow = Wrh + (size_t)col * Hh;
#pragma unroll
    for (int kb = 0; kb < 4; ++kb) acc = wmma_f16(afr[kb], load_b_h(Wrow + kb * 32, gsel), acc);
    float bias = br[col];
    int rb = row0 + (gsel << 3);
#pragma unroll
    for (int i = 0; i < 8; ++i) out[(size_t)(rb + i) * KGEc + col] = acc[i] + bias;
  }
}

// attn[g] = sigmoid(prelu((gsum[g]/cnt[g]) @ Wa1^T + ba1, a2) @ Wa2^T + ba2); tiny: 64 thr/graph
__global__ void k_attn(const float* __restrict__ gsum, const float* __restrict__ cntg,
                       const float* __restrict__ Wa1, const float* __restrict__ ba1,
                       const float* __restrict__ a2, const float* __restrict__ Wa2,
                       const float* __restrict__ ba2, float* __restrict__ attn) {
  __shared__ float sh[64];
  int g = blockIdx.x;
  int u = threadIdx.x;
  float inv = 1.0f / fmaxf(cntg[g], 1.0f);
  const float* gr = gsum + (size_t)g * Hh;
  const float* w = Wa1 + (size_t)u * Hh;
  float s = 0.0f;
#pragma unroll 4
  for (int k = 0; k < Hh; ++k) s += gr[k] * w[k];
  s = s * inv + ba1[u];
  float al = a2[0];
  s = (s >= 0.0f) ? s : al * s;
  sh[u] = s * Wa2[u];
  __syncthreads();
  for (int o = 32; o > 0; o >>= 1) {
    if (u < o) sh[u] += sh[u + o];
    __syncthreads();
  }
  if (u == 0) attn[g] = sigmf(sh[0] + ba2[0]);
}

// ---------------- host launcher ----------------

extern "C" void kernel_launch(void* const* d_in, const int* in_sizes, int n_in,
                              void* d_out, int out_size, void* d_ws, size_t ws_size,
                              hipStream_t stream) {
  const float* x    = (const float*)d_in[0];
  const float* ea   = (const float*)d_in[1];
  const int*   ei   = (const int*)d_in[2];
  const int*   bat  = (const int*)d_in[3];
  const int*   lgei = (const int*)d_in[4];
  const float* W1   = (const float*)d_in[5];
  const float* b1   = (const float*)d_in[6];
  const float* a1   = (const float*)d_in[7];
  const float* bng  = (const float*)d_in[8];
  const float* bnb  = (const float*)d_in[9];
  const float* bnm  = (const float*)d_in[10];
  const float* bnv  = (const float*)d_in[11];
  const float* wih  = (const float*)d_in[12];
  const float* whh  = (const float*)d_in[13];
  const float* bih  = (const float*)d_in[14];
  const float* bhh  = (const float*)d_in[15];
  const float* Wa1  = (const float*)d_in[16];
  const float* ba1  = (const float*)d_in[17];
  const float* a2   = (const float*)d_in[18];
  const float* Wa2  = (const float*)d_in[19];
  const float* ba2  = (const float*)d_in[20];
  const float* Wr   = (const float*)d_in[21];
  const float* br   = (const float*)d_in[22];

  const int* src  = ei;
  const int* dst  = ei + Ee;
  const int* lsrc = lgei;
  const int* ldst = lgei + ELG;

  char* ws = (char*)d_ws;
  size_t off = 0;
  auto alloc = [&](size_t bytes) -> void* {
    void* p = ws + off;
    off += (bytes + 255) & ~(size_t)255;
    return p;
  };
  float* fused = (float*)alloc((size_t)Ee * Hh * 4);
  float* agg   = (float*)alloc((size_t)Ee * Hh * 4);
  float* nupd  = (float*)alloc((size_t)Nn * Hh * 4);
  float* deglg = (float*)alloc((size_t)Ee * 4);
  float* degn  = (float*)alloc((size_t)Nn * 4);
  float* cntg  = (float*)alloc((size_t)NG * 4);
  float* gsum  = (float*)alloc((size_t)NG * Hh * 4);
  float* attn  = (float*)alloc((size_t)NG * 4);
  _Float16* W1h  = (_Float16*)alloc((size_t)Hh * Hh * 2);
  _Float16* wihh = (_Float16*)alloc((size_t)3 * Hh * Hh * 2);
  _Float16* whhh = (_Float16*)alloc((size_t)3 * Hh * Hh * 2);
  _Float16* Wrh  = (_Float16*)alloc((size_t)KGEc * Hh * 2);

  float* h   = (float*)d_out;                       // x output region doubles as GRU state
  float* emb = (float*)d_out + (size_t)Nn * Hh;

  hipMemcpyAsync(h, x, (size_t)Nn * Hh * 4, hipMemcpyDeviceToDevice, stream);
  hipMemsetAsync(deglg, 0, (size_t)Ee * 4, stream);
  hipMemsetAsync(degn, 0, (size_t)Nn * 4, stream);
  hipMemsetAsync(cntg, 0, (size_t)NG * 4, stream);
  hipMemsetAsync(gsum, 0, (size_t)NG * Hh * 4, stream);

  k_cvt<<<(Hh * Hh + 255) / 256, 256, 0, stream>>>(W1, W1h, Hh * Hh);
  k_cvt<<<(3 * Hh * Hh + 255) / 256, 256, 0, stream>>>(wih, wihh, 3 * Hh * Hh);
  k_cvt<<<(3 * Hh * Hh + 255) / 256, 256, 0, stream>>>(whh, whhh, 3 * Hh * Hh);
  k_cvt<<<(KGEc * Hh + 255) / 256, 256, 0, stream>>>(Wr, Wrh, KGEc * Hh);

  k_count<<<(ELG + 255) / 256, 256, 0, stream>>>(ldst, deglg, ELG);
  k_count<<<(Ee + 255) / 256, 256, 0, stream>>>(dst, degn, Ee);
  k_count<<<(Nn + 255) / 256, 256, 0, stream>>>(bat, cntg, Nn);

  for (int it = 0; it < NITER; ++it) {
    k_fuse<<<(Ee * 32 + 255) / 256, 256, 0, stream>>>(ea, h, src, dst, fused);
    hipMemsetAsync(agg, 0, (size_t)Ee * Hh * 4, stream);
    k_scatter_add<<<(ELG * 32 + 255) / 256, 256, 0, stream>>>(fused, lsrc, ldst, agg, ELG);
    k_edge_gemm<<<((Ee / 16) * 32 + 255) / 256, 256, 0, stream>>>(agg, deglg, W1h, b1, a1,
                                                                  bng, bnb, bnm, bnv, fused);
    hipMemsetAsync(nupd, 0, (size_t)Nn * Hh * 4, stream);
    k_scatter_add<<<(Ee * 32 + 255) / 256, 256, 0, stream>>>(fused, nullptr, dst, nupd, Ee);
    k_gru<<<((Nn / 16) * 32 + 255) / 256, 256, 0, stream>>>(nupd, degn, wihh, whhh, bih, bhh, h);
  }

  k_scatter_add<<<(Nn * 32 + 255) / 256, 256, 0, stream>>>(h, nullptr, bat, gsum, Nn);
  k_attn<<<NG, 64, 0, stream>>>(gsum, cntg, Wa1, ba1, a2, Wa2, ba2, attn);
  k_emb<<<((NG / 16) * 32 + 255) / 256, 256, 0, stream>>>(gsum, attn, Wrh, br, emb);
}
